// GNN_46179488366795
// MI455X (gfx1250) — matrix operand order repaired and checked by reading, not compiled, not run on hardware
//
#include <hip/hip_runtime.h>
#include <hip/hip_bf16.h>
#include <math.h>

// ---------------------------------------------------------------------------
// 2-layer GCN: GEMM(+bias) -> SpMM -> ReLU -> GEMM(+bias) -> SpMM -> log_softmax
// GEMMs use v_wmma_f32_16x16x32_bf16 with split-bf16 (hi+lo) for ~fp32 accuracy.
// Weights are pre-transposed + pre-split to bf16 once per launch so the GEMM
// inner loop does pure b128 loads + WMMA (no per-use conversion VALU for B).
// ---------------------------------------------------------------------------

typedef __attribute__((ext_vector_type(16))) __bf16 v16bf;
typedef __attribute__((ext_vector_type(8)))  __bf16 v8bf;
typedef __attribute__((ext_vector_type(8)))  float  v8f;

#define NNODES 50000
#define FDIM   512
#define HDIM   256
#define CDIM   64

// ---------------------------------------------------------------------------
// Weight prep: W[K][NC] fp32 -> W^T hi/lo bf16 [NC][K].
// hi = bf16(x), lo = bf16(x - float(hi));  hi+lo recovers ~fp32 precision.
// ---------------------------------------------------------------------------
template<int K, int NC>
__global__ __launch_bounds__(256)
void weight_splitT(const float* __restrict__ W,
                   __bf16* __restrict__ hiT, __bf16* __restrict__ loT)
{
    int idx = blockIdx.x * blockDim.x + threadIdx.x;
    if (idx >= K * NC) return;
    int n = idx / K;
    int k = idx - n * K;
    float x = W[(size_t)k * NC + n];
    __bf16 h = (__bf16)x;
    hiT[idx] = h;
    loT[idx] = (__bf16)(x - (float)h);
}

// ---------------------------------------------------------------------------
// GEMM: C[M,NCOLS] = A[M,KTOT] @ B[KTOT,NCOLS] + bias, optional ReLU on A.
// One wave computes a 16x64 slab of C (4 x 16x16 WMMA accumulators).
// B supplied pre-split/pre-transposed bf16: per-lane fragment is contiguous
// K=16*hl..16*hl+15 -> two global_load_b128, zero conversion VALU.
// A split to bf16 hi/lo on the fly (reused across 4 N-tiles, co-executes
// with the XDL-tracked WMMAs).
// ---------------------------------------------------------------------------
template<int KTOT, int NCOLS, bool RELU>
__global__ __launch_bounds__(256)
void gemm_bias_wmma(const float* __restrict__ A,
                    const __bf16* __restrict__ BhiT,
                    const __bf16* __restrict__ BloT,
                    const float* __restrict__ bias,
                    float* __restrict__ C, int M)
{
    constexpr int NGROUPS = NCOLS / 64;       // 64-wide slabs per row of tiles
    const int tilesM = M >> 4;
    const int wave   = (blockIdx.x * blockDim.x + threadIdx.x) >> 5;
    const int tile_m = wave / NGROUPS;
    const int ng     = wave % NGROUPS;
    if (tile_m >= tilesM) return;             // wave-uniform exit (EXEC stays full)

    const int lane = threadIdx.x & 31;
    const int hl   = lane >> 4;               // lane half (0: lanes 0-15, 1: 16-31)
    const int mr   = lane & 15;               // A-row / B-col within tile
    const int n_base = ng * 64;

    v8f acc[4] = {};

    const float*  arow = A + (size_t)(tile_m * 16 + mr) * KTOT;
    const __bf16* bh0  = BhiT + (size_t)(n_base + mr) * KTOT + 16 * hl;
    const __bf16* bl0  = BloT + (size_t)(n_base + mr) * KTOT + 16 * hl;

    for (int k0 = 0; k0 < KTOT; k0 += 32) {
        // ---- A fragment (16x32 bf16): lanes 0-15 K=0-15, 16-31 K=16-31
        v16bf ahi, alo;
#pragma unroll
        for (int r = 0; r < 8; ++r) {
            int kk = k0 + 2 * r + 16 * hl;
            float2 a2 = *(const float2*)(arow + kk);
            float x0 = a2.x, x1 = a2.y;
            if (RELU) { x0 = fmaxf(x0, 0.0f); x1 = fmaxf(x1, 0.0f); }
            __bf16 h0 = (__bf16)x0, h1 = (__bf16)x1;
            ahi[2*r]   = h0;
            ahi[2*r+1] = h1;
            alo[2*r]   = (__bf16)(x0 - (float)h0);
            alo[2*r+1] = (__bf16)(x1 - (float)h1);
        }

#pragma unroll
        for (int nt = 0; nt < 4; ++nt) {
            const size_t boff = (size_t)nt * 16 * KTOT + k0;
            v8bf h0 = *(const v8bf*)(bh0 + boff);
            v8bf h1 = *(const v8bf*)(bh0 + boff + 8);
            v8bf l0 = *(const v8bf*)(bl0 + boff);
            v8bf l1 = *(const v8bf*)(bl0 + boff + 8);
            v16bf bhi = __builtin_shufflevector(h0, h1,
                0,1,2,3,4,5,6,7,8,9,10,11,12,13,14,15);
            v16bf blo = __builtin_shufflevector(l0, l1,
                0,1,2,3,4,5,6,7,8,9,10,11,12,13,14,15);

            acc[nt] = __builtin_amdgcn_wmma_f32_16x16x32_bf16(
                false, ahi, false, bhi, (short)0, acc[nt], false, false);
            acc[nt] = __builtin_amdgcn_wmma_f32_16x16x32_bf16(
                false, alo, false, bhi, (short)0, acc[nt], false, false);
            acc[nt] = __builtin_amdgcn_wmma_f32_16x16x32_bf16(
                false, ahi, false, blo, (short)0, acc[nt], false, false);
        }
    }

    // ---- Epilogue: C/D layout — VGPR r -> row (r + 8*half), col = lane&15
#pragma unroll
    for (int nt = 0; nt < 4; ++nt) {
        int n = n_base + nt * 16 + mr;
        float bb = bias[n];
#pragma unroll
        for (int r = 0; r < 8; ++r) {
            int m = tile_m * 16 + r + 8 * hl;
            C[(size_t)m * NCOLS + n] = acc[nt][r] + bb;
        }
    }
}

// ---------------------------------------------------------------------------
// SpMM scatter: out[rows[e],:] += vals[e] * dense[cols[e],:]
// One thread per (edge, 4-float chunk); hardware f32 atomics resolve in L2
// (operands are L2-resident: 51 MB / 13 MB buffers vs 192 MB L2).
// ---------------------------------------------------------------------------
template<int D>
__global__ __launch_bounds__(256)
void spmm_scatter(const int* __restrict__ rows, const int* __restrict__ cols,
                  const float* __restrict__ vals, const float* __restrict__ dense,
                  float* __restrict__ out, int E)
{
    constexpr int TPE = D / 4;                 // threads per edge
    long long idx = (long long)blockIdx.x * blockDim.x + threadIdx.x;
    int e = (int)(idx / TPE);
    if (e >= E) return;
    int f = (int)(idx % TPE) * 4;

    int   ci = cols[e];
    int   ri = rows[e];
    float v  = vals[e];

    const float4 d = *(const float4*)(dense + (size_t)ci * D + f);
    float* o = out + (size_t)ri * D + f;
    unsafeAtomicAdd(o + 0, v * d.x);
    unsafeAtomicAdd(o + 1, v * d.y);
    unsafeAtomicAdd(o + 2, v * d.z);
    unsafeAtomicAdd(o + 3, v * d.w);
}

// ---------------------------------------------------------------------------
// log_softmax over rows of 64: one wave32 per row, 2 elements/lane,
// max + sum-exp reductions via __shfl_xor.
// ---------------------------------------------------------------------------
__global__ __launch_bounds__(256)
void log_softmax64(const float* __restrict__ y, float* __restrict__ out, int N)
{
    int row = (blockIdx.x * blockDim.x + threadIdx.x) >> 5;
    if (row >= N) return;
    int lane = threadIdx.x & 31;

    float2 x = *(const float2*)(y + (size_t)row * 64 + lane * 2);

    float m = fmaxf(x.x, x.y);
#pragma unroll
    for (int off = 16; off > 0; off >>= 1)
        m = fmaxf(m, __shfl_xor(m, off, 32));

    float s = __expf(x.x - m) + __expf(x.y - m);
#pragma unroll
    for (int off = 16; off > 0; off >>= 1)
        s += __shfl_xor(s, off, 32);

    float lse = m + __logf(s);
    float2 r;
    r.x = x.x - lse;
    r.y = x.y - lse;
    *(float2*)(out + (size_t)row * 64 + lane * 2) = r;
}

// ---------------------------------------------------------------------------
extern "C" void kernel_launch(void* const* d_in, const int* in_sizes, int n_in,
                              void* d_out, int out_size, void* d_ws, size_t ws_size,
                              hipStream_t stream)
{
    const float* X    = (const float*)d_in[0];   // [N, F]
    const int*   rows = (const int*)  d_in[1];   // [E]
    const int*   cols = (const int*)  d_in[2];   // [E]
    const float* vals = (const float*)d_in[3];   // [E]
    const float* W1   = (const float*)d_in[4];   // [F, H]
    const float* b1   = (const float*)d_in[5];   // [H]
    const float* W2   = (const float*)d_in[6];   // [H, C]
    const float* b2   = (const float*)d_in[7];   // [C]
    float* out = (float*)d_out;                  // [N, C]

    const int N = NNODES;
    const int E = in_sizes[1];

    // Workspace layout: fp32 intermediates (128 MB) + bf16 weight splits (~0.6 MB)
    float* support1 = (float*)d_ws;                  // N*H
    float* xagg     = support1 + (size_t)N * HDIM;   // N*H
    float* support2 = xagg     + (size_t)N * HDIM;   // N*C
    float* yagg     = support2 + (size_t)N * CDIM;   // N*C
    __bf16* W1hiT   = (__bf16*)(yagg + (size_t)N * CDIM);  // [H][F]
    __bf16* W1loT   = W1hiT + (size_t)FDIM * HDIM;
    __bf16* W2hiT   = W1loT + (size_t)FDIM * HDIM;         // [C][H]
    __bf16* W2loT   = W2hiT + (size_t)HDIM * CDIM;

    hipMemsetAsync(xagg, 0, (size_t)N * HDIM * sizeof(float), stream);
    hipMemsetAsync(yagg, 0, (size_t)N * CDIM * sizeof(float), stream);

    {   // Weight prep: transpose + split to bf16 hi/lo (once per launch)
        int n1 = FDIM * HDIM, n2 = HDIM * CDIM;
        weight_splitT<FDIM, HDIM><<<(n1 + 255) / 256, 256, 0, stream>>>(W1, W1hiT, W1loT);
        weight_splitT<HDIM, CDIM><<<(n2 + 255) / 256, 256, 0, stream>>>(W2, W2hiT, W2loT);
    }
    {   // Layer 1 GEMM: support1 = X @ W1 + b1
        int waves  = (N / 16) * (HDIM / 64);         // 3125 * 4
        int blocks = (waves + 7) / 8;
        gemm_bias_wmma<FDIM, HDIM, false><<<blocks, 256, 0, stream>>>(
            X, W1hiT, W1loT, b1, support1, N);
    }
    {   // SpMM 1: xagg += vals * support1[cols]
        long long total = (long long)E * (HDIM / 4);
        int blocks = (int)((total + 255) / 256);
        spmm_scatter<HDIM><<<blocks, 256, 0, stream>>>(
            rows, cols, vals, support1, xagg, E);
    }
    {   // Layer 2 GEMM with fused ReLU on A: support2 = relu(xagg) @ W2 + b2
        int waves  = (N / 16) * (CDIM / 64);         // 3125
        int blocks = (waves + 7) / 8;
        gemm_bias_wmma<HDIM, CDIM, true><<<blocks, 256, 0, stream>>>(
            xagg, W2hiT, W2loT, b2, support2, N);
    }
    {   // SpMM 2: yagg += vals * support2[cols]
        long long total = (long long)E * (CDIM / 4);
        int blocks = (int)((total + 255) / 256);
        spmm_scatter<CDIM><<<blocks, 256, 0, stream>>>(
            rows, cols, vals, support2, yagg, E);
    }
    {   // log_softmax rows
        int threads = N * 32;
        int blocks  = (threads + 255) / 256;
        log_softmax64<<<blocks, 256, 0, stream>>>(yagg, out, N);
    }
}
